// MultiHeadAttention_25048249270546
// MI455X (gfx1250) — compile-verified, matrix operand order checked by reading, not run on hardware
//
#include <hip/hip_runtime.h>
#include <hip/hip_bf16.h>
#include <math.h>

// ---------------------------------------------------------------------------
// Problem constants (reference: b=2, s=2048, d=2048, 32 q-heads, 8 kv-heads,
// head_dim=64).
// ---------------------------------------------------------------------------
#define BATCH     2
#define SEQ       2048
#define DMODEL    2048
#define NH        32
#define NKVH      8
#define HD        64
#define KVDIM     (NKVH * HD)      // 512
#define TOKENS    (BATCH * SEQ)    // 4096

// LDS row padding (halves) for conflict-free strided row access
#define LPAD      40

typedef __attribute__((ext_vector_type(16))) _Float16 v16h;
typedef __attribute__((ext_vector_type(8)))  _Float16 v8h;
typedef __attribute__((ext_vector_type(8)))  float    v8f;
typedef int v4i_vec __attribute__((vector_size(4 * sizeof(int))));

// Async global->LDS path (gfx1250 ASYNCcnt); guarded so either toolchain
// compiles. Fallback = synchronous load + ds_store staging.
#if defined(__has_builtin)
#  if __has_builtin(__builtin_amdgcn_global_load_async_to_lds_b128) && \
      __has_builtin(__builtin_amdgcn_s_wait_asynccnt)
#    define HAVE_ASYNC 1
#  endif
#endif
#ifndef HAVE_ASYNC
#  define HAVE_ASYNC 0
#endif

#if HAVE_ASYNC
__device__ __forceinline__ void async_copy_b128(const void* g, void* l) {
  __builtin_amdgcn_global_load_async_to_lds_b128(
      (__attribute__((address_space(1))) v4i_vec*)g,
      (__attribute__((address_space(3))) v4i_vec*)l, 0, 0);
}
#endif

// ---------------------------------------------------------------------------
// Build a 16-half WMMA fragment for one lane from a row pointer (global or
// LDS). CDNA5 16-bit A-layout (B mirrors it): lane holds row (lane&15);
// halves 0..7 -> k = kg+0..7, halves 8..15 -> k = 16+kg+0..7, kg=(lane>>4)*8.
// ---------------------------------------------------------------------------
__device__ __forceinline__ v16h load_frag_h(const _Float16* p, int kg) {
  v8h lo = *reinterpret_cast<const v8h*>(p + kg);
  v8h hi = *reinterpret_cast<const v8h*>(p + 16 + kg);
  v16h f;
#pragma unroll
  for (int t = 0; t < 8; ++t) { f[t] = lo[t]; f[8 + t] = hi[t]; }
  return f;
}

__device__ __forceinline__ v8f wmma32(v16h a, v16h b, v8f c) {
  return __builtin_amdgcn_wmma_f32_16x16x32_f16(false, a, false, b, (short)0,
                                                c, false, false);
}

// ---------------------------------------------------------------------------
// Elementwise f32 -> f16 conversion (4 elements/thread; n % 4 == 0).
// ---------------------------------------------------------------------------
__global__ __launch_bounds__(256) void cvt_f32_f16_kernel(
    const float* __restrict__ in, _Float16* __restrict__ out, long n) {
  const long i4 = ((long)blockIdx.x * blockDim.x + threadIdx.x) * 4;
  if (i4 >= n) return;
  float4 v = *reinterpret_cast<const float4*>(in + i4);
  out[i4 + 0] = (_Float16)v.x;
  out[i4 + 1] = (_Float16)v.y;
  out[i4 + 2] = (_Float16)v.z;
  out[i4 + 3] = (_Float16)v.w;
}

// ---------------------------------------------------------------------------
// C[M,N] = A[M,K] @ W[N,K]^T ; f16 inputs, f32 WMMA accumulate.
// Block tile 64(M) x 256(N); 8 waves as 2x4, each wave = 32x64 (8 WMMA/step).
// A/B 32-wide k-slices double-buffered in LDS via async global->LDS copies.
// grid = (N/256, M/64).  F16OUT selects f16 vs f32 output.
// ---------------------------------------------------------------------------
template <bool F16OUT>
__global__ __launch_bounds__(256) void gemm_h_kernel(
    const _Float16* __restrict__ A, const _Float16* __restrict__ W,
    float* __restrict__ Cf, _Float16* __restrict__ Ch, int M, int N, int K) {
  const int t    = threadIdx.x;
  const int wave = t >> 5;
  const int lane = t & 31;
  const int wr   = wave >> 2;        // 0..1  (M sub-tile)
  const int wc   = wave & 3;         // 0..3  (N sub-tile)
  const int r    = lane & 15;
  const int kg   = (lane >> 4) * 8;

  const int gm0 = blockIdx.y * 64;   // block rows in A
  const int gn0 = blockIdx.x * 256;  // block rows in W

  __shared__ __align__(16) _Float16 As[2][64][LPAD];
  __shared__ __align__(16) _Float16 Bs[2][256][LPAD];

  // cooperative stage loader: 256 threads move A(64x32) + B(256x32) halves
  const int srow = t >> 2;           // 0..63
  const int sseg = (t & 3) * 8;      // 0,8,16,24 (halves)

  auto stage_load = [&](int kk, int s) {
    const _Float16* ga = A + (size_t)(gm0 + srow) * K + kk + sseg;
#if HAVE_ASYNC
    async_copy_b128(ga, &As[s][srow][sseg]);
#else
    *reinterpret_cast<v8h*>(&As[s][srow][sseg]) =
        *reinterpret_cast<const v8h*>(ga);
#endif
#pragma unroll
    for (int j = 0; j < 4; ++j) {
      const int brow = srow + 64 * j;
      const _Float16* gb = W + (size_t)(gn0 + brow) * K + kk + sseg;
#if HAVE_ASYNC
      async_copy_b128(gb, &Bs[s][brow][sseg]);
#else
      *reinterpret_cast<v8h*>(&Bs[s][brow][sseg]) =
          *reinterpret_cast<const v8h*>(gb);
#endif
    }
  };

  v8f acc[2][4];
#pragma unroll
  for (int a = 0; a < 2; ++a)
#pragma unroll
    for (int c = 0; c < 4; ++c) acc[a][c] = (v8f){};

  stage_load(0, 0);

  int s = 0;
  for (int kk = 0; kk < K; kk += 32, s ^= 1) {
#if HAVE_ASYNC
    __builtin_amdgcn_s_wait_asynccnt(0);
#endif
    __syncthreads();  // stage s resident; all waves past reads of buffer s^1

    if (kk + 32 < K) stage_load(kk + 32, s ^ 1);

    // fragments from LDS (conflict-free: row stride = LPAD halves)
    const v16h af0 = load_frag_h(&As[s][wr * 32 + r][0], kg);
    const v16h af1 = load_frag_h(&As[s][wr * 32 + 16 + r][0], kg);
    const v16h bf0 = load_frag_h(&Bs[s][wc * 64 + r][0], kg);
    const v16h bf1 = load_frag_h(&Bs[s][wc * 64 + 16 + r][0], kg);
    const v16h bf2 = load_frag_h(&Bs[s][wc * 64 + 32 + r][0], kg);
    const v16h bf3 = load_frag_h(&Bs[s][wc * 64 + 48 + r][0], kg);

    acc[0][0] = wmma32(af0, bf0, acc[0][0]);
    acc[0][1] = wmma32(af0, bf1, acc[0][1]);
    acc[0][2] = wmma32(af0, bf2, acc[0][2]);
    acc[0][3] = wmma32(af0, bf3, acc[0][3]);
    acc[1][0] = wmma32(af1, bf0, acc[1][0]);
    acc[1][1] = wmma32(af1, bf1, acc[1][1]);
    acc[1][2] = wmma32(af1, bf2, acc[1][2]);
    acc[1][3] = wmma32(af1, bf3, acc[1][3]);
  }

  const int m0 = gm0 + wr * 32;
  const int n0 = gn0 + wc * 64;
  const int mo = 8 * (lane >> 4);
#pragma unroll
  for (int a = 0; a < 2; ++a) {
#pragma unroll
    for (int c = 0; c < 4; ++c) {
#pragma unroll
      for (int i = 0; i < 8; ++i) {
        const size_t off = (size_t)(m0 + 16 * a + i + mo) * N + n0 + 16 * c + r;
        if (F16OUT) Ch[off] = (_Float16)acc[a][c][i];
        else        Cf[off] = acc[a][c][i];
      }
    }
  }
}

// ---------------------------------------------------------------------------
// In-place RoPE on f16 t: layout [TOKENS][nHeads*HD]; f32 math.
// `scale` folds the softmax 1/sqrt(HD) into Q.
// ---------------------------------------------------------------------------
__global__ __launch_bounds__(256) void rope_h_kernel(_Float16* __restrict__ t,
                                                     int nHeads, float scale) {
  const long idx = (long)blockIdx.x * blockDim.x + threadIdx.x;
  const long total = (long)TOKENS * nHeads * (HD / 2);
  if (idx >= total) return;
  const int  i   = (int)(idx % (HD / 2));
  long       tmp = idx / (HD / 2);
  const int  hh  = (int)(tmp % nHeads);
  const long tok = tmp / nHeads;
  const int  pos = (int)(tok % SEQ);

  const float inv = __expf(-(float)(2 * i) * (9.210340371976184f / (float)HD));
  const float ang = (float)pos * inv;
  float sn, cs;
  __sincosf(ang, &sn, &cs);

  _Float16* p = t + tok * ((long)nHeads * HD) + hh * HD + 2 * i;
  const float t1 = (float)p[0], t2 = (float)p[1];
  p[0] = (_Float16)((t1 * cs - t2 * sn) * scale);
  p[1] = (_Float16)((t1 * sn + t2 * cs) * scale);
}

// ---------------------------------------------------------------------------
// vt[b][kvh][d][s] = v[b][s][kvh*64+d]  (f16; contiguous along s for k-loads)
// ---------------------------------------------------------------------------
__global__ __launch_bounds__(256) void transpose_v_kernel(
    const _Float16* __restrict__ v, _Float16* __restrict__ vt) {
  const long idx = (long)blockIdx.x * blockDim.x + threadIdx.x;
  const long total = (long)TOKENS * KVDIM;
  if (idx >= total) return;
  const int  d   = (int)(idx % KVDIM);
  const long tok = idx / KVDIM;
  const int  b   = (int)(tok / SEQ);
  const int  sidx = (int)(tok % SEQ);
  const int  kvh = d >> 6;
  const int  dd  = d & 63;
  vt[(((size_t)(b * NKVH + kvh) * HD) + dd) * SEQ + sidx] = v[idx];
}

// ---------------------------------------------------------------------------
// Flash-style causal GQA attention, all-f16 operands, f32 online softmax.
//   q  : f16 [b][s][NH*HD]   (RoPE + 1/sqrt(HD) applied)
//   k  : f16 [b][s][NKVH*HD] (RoPE applied)
//   vt : f16 [b][kvh][HD][s]
//   att: f16 [b][s][NH*HD]
// One wave per (b, h, 16-row q tile); P re-laid-out via LDS.
// ---------------------------------------------------------------------------
__global__ __launch_bounds__(32) void attn_kernel(
    const _Float16* __restrict__ q, const _Float16* __restrict__ k,
    const _Float16* __restrict__ vt, _Float16* __restrict__ att) {
  const int qt = blockIdx.x;
  const int h  = blockIdx.y;
  const int b  = blockIdx.z;
  const int lane = threadIdx.x;
  const int r  = lane & 15;
  const int hi = lane >> 4;
  const int kg = hi * 8;
  const int q0 = qt * 16;
  const int kvh = h >> 2;  // NH/NKVH = 4

  const _Float16* qrow = q + (size_t)(b * SEQ + q0 + r) * DMODEL + h * HD;
  const v16h qa0 = load_frag_h(qrow, kg);
  const v16h qa1 = load_frag_h(qrow + 32, kg);

  const _Float16* kbase = k + (size_t)b * SEQ * KVDIM + kvh * HD;
  const _Float16* vbase = vt + ((size_t)(b * NKVH + kvh) * HD) * SEQ;

  v8f oacc[4];
  float mrow[8], lrow[8];
#pragma unroll
  for (int c = 0; c < 4; ++c) oacc[c] = (v8f){};
#pragma unroll
  for (int i = 0; i < 8; ++i) { mrow[i] = -3.0e38f; lrow[i] = 0.0f; }

  __shared__ __align__(16) _Float16 P[16][32];

  const int kvEnd = q0 + 16;
  for (int kv0 = 0; kv0 < kvEnd; kv0 += 32) {
    // --- S = Q * K^T for kv columns kv0..kv0+31 (k = 64, 2 chained WMMAs) ---
    const _Float16* kr0 = kbase + (size_t)(kv0 + r) * KVDIM;
    const _Float16* kr1 = kbase + (size_t)(kv0 + 16 + r) * KVDIM;
    v8f c0 = {};
    c0 = wmma32(qa0, load_frag_h(kr0, kg), c0);
    c0 = wmma32(qa1, load_frag_h(kr0 + 32, kg), c0);
    v8f c1 = {};
    c1 = wmma32(qa0, load_frag_h(kr1, kg), c1);
    c1 = wmma32(qa1, load_frag_h(kr1 + 32, kg), c1);

    // --- causal mask + online softmax (row stats via half-wave shuffles) ---
#pragma unroll
    for (int i = 0; i < 8; ++i) {
      const int row  = q0 + i + 8 * hi;
      const int col0 = kv0 + r;
      const int col1 = kv0 + 16 + r;
      float s0 = (col0 <= row) ? c0[i] : -3.0e38f;
      float s1 = (col1 <= row) ? c1[i] : -3.0e38f;

      float mx = fmaxf(s0, s1);
#pragma unroll
      for (int d = 1; d < 16; d <<= 1) mx = fmaxf(mx, __shfl_xor(mx, d, 32));
      const float mnew = fmaxf(mrow[i], mx);

      const float p0 = __expf(s0 - mnew);
      const float p1 = __expf(s1 - mnew);
      float sum = p0 + p1;
#pragma unroll
      for (int d = 1; d < 16; d <<= 1) sum += __shfl_xor(sum, d, 32);

      const float sc = __expf(mrow[i] - mnew);
      lrow[i] = lrow[i] * sc + sum;
      mrow[i] = mnew;
#pragma unroll
      for (int c = 0; c < 4; ++c) oacc[c][i] *= sc;

      P[i + 8 * hi][r]      = (_Float16)p0;
      P[i + 8 * hi][16 + r] = (_Float16)p1;
    }
    __syncthreads();  // order LDS stores vs. A-layout reloads

    // --- reload P in A-layout (two 16B LDS vector reads) ---
    v8h plo = *reinterpret_cast<const v8h*>(&P[r][kg]);
    v8h phi = *reinterpret_cast<const v8h*>(&P[r][16 + kg]);
    v16h pa;
#pragma unroll
    for (int u = 0; u < 8; ++u) { pa[u] = plo[u]; pa[8 + u] = phi[u]; }
    __syncthreads();

    // --- O += P * V  (d in four 16-wide chunks) ---
#pragma unroll
    for (int c = 0; c < 4; ++c) {
      const _Float16* vr = vbase + (size_t)(c * 16 + r) * SEQ + kv0;
      oacc[c] = wmma32(pa, load_frag_h(vr, kg), oacc[c]);
    }
  }

  // --- normalize and write f16 [b][s][NH*HD] ---
  _Float16* obase = att + (size_t)b * SEQ * DMODEL + h * HD;
#pragma unroll
  for (int c = 0; c < 4; ++c) {
#pragma unroll
    for (int i = 0; i < 8; ++i) {
      const int row = q0 + i + 8 * hi;
      obase[(size_t)row * DMODEL + c * 16 + r] = (_Float16)(oacc[c][i] / lrow[i]);
    }
  }
}

// ---------------------------------------------------------------------------
// Host launcher. Inputs: x, wq, wk, wv, wo (f32). Output: f32 [b][s][d].
// ---------------------------------------------------------------------------
extern "C" void kernel_launch(void* const* d_in, const int* in_sizes, int n_in,
                              void* d_out, int out_size, void* d_ws,
                              size_t ws_size, hipStream_t stream) {
  (void)in_sizes; (void)n_in; (void)out_size; (void)ws_size;

  const float* x  = (const float*)d_in[0];
  const float* wq = (const float*)d_in[1];
  const float* wk = (const float*)d_in[2];
  const float* wv = (const float*)d_in[3];
  const float* wo = (const float*)d_in[4];
  float* out = (float*)d_out;

  // f16 workspace layout
  _Float16* xh  = (_Float16*)d_ws;                     // TOKENS*DMODEL
  _Float16* wqh = xh  + (size_t)TOKENS * DMODEL;       // DMODEL*DMODEL
  _Float16* wkh = wqh + (size_t)DMODEL * DMODEL;       // KVDIM*DMODEL
  _Float16* wvh = wkh + (size_t)KVDIM * DMODEL;        // KVDIM*DMODEL
  _Float16* woh = wvh + (size_t)KVDIM * DMODEL;        // DMODEL*DMODEL
  _Float16* qh  = woh + (size_t)DMODEL * DMODEL;       // TOKENS*DMODEL
  _Float16* kh  = qh  + (size_t)TOKENS * DMODEL;       // TOKENS*KVDIM
  _Float16* vh  = kh  + (size_t)TOKENS * KVDIM;        // TOKENS*KVDIM
  _Float16* vth = vh  + (size_t)TOKENS * KVDIM;        // TOKENS*KVDIM
  _Float16* ath = vth + (size_t)TOKENS * KVDIM;        // TOKENS*DMODEL

  const dim3 blk256(256);
  auto grid1d = [](long n) { return dim3((unsigned)((n + 1023) / 1024)); };

  // Convert inputs to f16 once
  cvt_f32_f16_kernel<<<grid1d((long)TOKENS * DMODEL), blk256, 0, stream>>>(
      x, xh, (long)TOKENS * DMODEL);
  cvt_f32_f16_kernel<<<grid1d((long)DMODEL * DMODEL), blk256, 0, stream>>>(
      wq, wqh, (long)DMODEL * DMODEL);
  cvt_f32_f16_kernel<<<grid1d((long)KVDIM * DMODEL), blk256, 0, stream>>>(
      wk, wkh, (long)KVDIM * DMODEL);
  cvt_f32_f16_kernel<<<grid1d((long)KVDIM * DMODEL), blk256, 0, stream>>>(
      wv, wvh, (long)KVDIM * DMODEL);
  cvt_f32_f16_kernel<<<grid1d((long)DMODEL * DMODEL), blk256, 0, stream>>>(
      wo, woh, (long)DMODEL * DMODEL);

  // QKV projections (f16 in, f16 out)
  gemm_h_kernel<true><<<dim3(DMODEL / 256, TOKENS / 64), blk256, 0, stream>>>(
      xh, wqh, nullptr, qh, TOKENS, DMODEL, DMODEL);
  gemm_h_kernel<true><<<dim3(KVDIM / 256, TOKENS / 64), blk256, 0, stream>>>(
      xh, wkh, nullptr, kh, TOKENS, KVDIM, DMODEL);
  gemm_h_kernel<true><<<dim3(KVDIM / 256, TOKENS / 64), blk256, 0, stream>>>(
      xh, wvh, nullptr, vh, TOKENS, KVDIM, DMODEL);

  // RoPE (scale folds softmax 1/sqrt(HD) into Q)
  {
    long nq = (long)TOKENS * NH * (HD / 2);
    rope_h_kernel<<<dim3((unsigned)((nq + 255) / 256)), blk256, 0, stream>>>(
        qh, NH, 0.125f);
    long nk = (long)TOKENS * NKVH * (HD / 2);
    rope_h_kernel<<<dim3((unsigned)((nk + 255) / 256)), blk256, 0, stream>>>(
        kh, NKVH, 1.0f);
  }

  // V transpose
  {
    long nv = (long)TOKENS * KVDIM;
    transpose_v_kernel<<<dim3((unsigned)((nv + 255) / 256)), blk256, 0, stream>>>(
        vh, vth);
  }

  // Causal GQA flash attention
  attn_kernel<<<dim3(SEQ / 16, NH, BATCH), dim3(32), 0, stream>>>(qh, kh, vth, ath);

  // Output projection (f16 in, f32 out)
  gemm_h_kernel<false><<<dim3(DMODEL / 256, TOKENS / 64), blk256, 0, stream>>>(
      ath, woh, out, nullptr, TOKENS, DMODEL, DMODEL);
}
